// SceneGraphGenerator_54331336295032
// MI455X (gfx1250) — compile-verified, matrix operand order checked by reading, not run on hardware
//
#include <hip/hip_runtime.h>
#include <hip/hip_bf16.h>
#include <math.h>

#define EPSF 1e-6f
constexpr int  kBt = 4, kN = 128, kD = 256, kH = 8, kT = 16;
constexpr long kE  = (long)kBt * kN * kN;   // 65536 edge rows
constexpr long kNB = (long)kBt * kN;        // 512 node rows

typedef __attribute__((ext_vector_type(16))) _Float16 v16h;
typedef __attribute__((ext_vector_type(8)))  _Float16 v8h;
typedef __attribute__((ext_vector_type(8)))  float    v8f;

enum { ACT_NONE = 0, ACT_GELU = 1, ACT_LEAKY = 2 };

__device__ __forceinline__ float apply_act(float v, int act) {
  if (act == ACT_GELU)  return 0.5f * v * (1.f + erff(v * 0.70710678118654752f));
  if (act == ACT_LEAKY) return v > 0.f ? v : 0.2f * v;
  return v;
}

// Force a load through the global (AS1) path -> global_load_b128, no DScnt.
template <typename T>
__device__ __forceinline__ T gload(const void* p) {
  return *(const __attribute__((address_space(1))) T*)p;
}

// ---------------------------------------------------------------------------
// WMMA GEMM. A row-major f16 (lda=K). Bs pre-swizzled to fragment order:
//   Bs[kt][ntile][lane][e]  (each lane's 16-half fragment contiguous = 32B)
// C (f32) / C16 (f16) both optional, row-major ldc=Nc.
// 128-thread block = 4 waves; each wave computes MT x NT 16x16 tiles.
// Requires: K % 32 == 0, M % (16*MT*4) == 0, Nc % (16*NT) == 0.
// ---------------------------------------------------------------------------
template <int MT, int NT>
__global__ __launch_bounds__(128) void gemm_wmma(
    const _Float16* __restrict__ A,
    const _Float16* __restrict__ Bs,
    const float* __restrict__ bias,
    float* __restrict__ C,
    _Float16* __restrict__ C16,
    int Nc, int K, long sA, long sB, long sC, int act) {
  const _Float16* Ab = A  + (long)blockIdx.z * sA;
  const _Float16* Bb = Bs + (long)blockIdx.z * sB;
  const int wv   = threadIdx.x >> 5;        // wave within workgroup
  const int lane = threadIdx.x & 31;
  const int half = lane >> 4, l16 = lane & 15;
  const long m0  = ((long)blockIdx.y * 4 + wv) * (16 * MT);
  const int  nt0 = blockIdx.x * NT;
  const int  ntiles = Nc >> 4;

  v8f acc[MT][NT];
#pragma unroll
  for (int mt = 0; mt < MT; ++mt)
#pragma unroll
    for (int nt = 0; nt < NT; ++nt) acc[mt][nt] = (v8f){};

  const _Float16* aptr = Ab + (m0 + l16) * (long)K + half * 8;
  const long arowoff = 16 * (long)K;            // A row-tile stride (elements)
  const _Float16* bptr = Bb + ((long)nt0 * 32 + lane) * 16;
  const long bstep = (long)ntiles * 512;

  for (int k0 = 0; k0 < K; k0 += 32) {
    v16h af[MT];
#pragma unroll
    for (int mt = 0; mt < MT; ++mt) {
      v8h lo = gload<v8h>(aptr + mt * arowoff);
      v8h hi = gload<v8h>(aptr + mt * arowoff + 16);
      af[mt] = __builtin_shufflevector(lo, hi, 0, 1, 2, 3, 4, 5, 6, 7,
                                       8, 9, 10, 11, 12, 13, 14, 15);
    }
    aptr += 32;
#pragma unroll
    for (int nt = 0; nt < NT; ++nt) {
      v16h bf = gload<v16h>(bptr + (long)nt * 512);
#pragma unroll
      for (int mt = 0; mt < MT; ++mt)
        acc[mt][nt] = __builtin_amdgcn_wmma_f32_16x16x32_f16(
            false, af[mt], false, bf, (short)0, acc[mt][nt], false, false);
    }
    bptr += bstep;
  }

#pragma unroll
  for (int nt = 0; nt < NT; ++nt) {
    const int n = (nt0 + nt) * 16 + l16;
    const float bsv = bias ? bias[n] : 0.f;
#pragma unroll
    for (int mt = 0; mt < MT; ++mt) {
#pragma unroll
      for (int r = 0; r < 8; ++r) {
        const long m = m0 + mt * 16 + r + 8 * half;
        const float v = apply_act(acc[mt][nt][r] + bsv, act);
        if (C)   (C   + (long)blockIdx.z * sC)[m * (long)Nc + n] = v;
        if (C16) (C16 + (long)blockIdx.z * sC)[m * (long)Nc + n] = (_Float16)v;
      }
    }
  }
}

// --- B-operand swizzle: f32 [R x C] (pad to Rp x Cp) -> fragment order ----
__global__ void cvtws_kernel(const float* __restrict__ src, _Float16* __restrict__ dst,
                             int R, int C, int Rp, int Cp, long sSrc, long sDst,
                             long total) {
  long idx = (long)blockIdx.x * blockDim.x + threadIdx.x;
  if (idx >= total) return;
  const long per = (long)(Rp >> 5) * (Cp >> 4) * 512;
  const int b = (int)(idx / per);
  long r = idx % per;
  const int e = (int)(r & 15);
  const int lane = (int)((r >> 4) & 31);
  long t = r >> 9;
  const int ntiles = Cp >> 4;
  const int nt = (int)(t % ntiles);
  const int kt = (int)(t / ntiles);
  const int half = lane >> 4, l16 = lane & 15;
  const int ko = e + ((e >> 3) << 3);
  const int k = kt * 32 + half * 8 + ko;
  const int n = nt * 16 + l16;
  dst[(long)b * sDst + r] =
      (k < R && n < C) ? (_Float16)src[(long)b * sSrc + (long)k * C + n] : (_Float16)0.f;
}

// --- plain f32 -> f16 convert (row-major, optional col pad) ---------------
__global__ void cvtk_kernel(const float* __restrict__ src, _Float16* __restrict__ dst,
                            long rows, int C, int Cp) {
  long idx = (long)blockIdx.x * blockDim.x + threadIdx.x;
  if (idx >= rows * (long)Cp) return;
  int c = (int)(idx % Cp);
  long r = idx / Cp;
  dst[idx] = (c < C) ? (_Float16)src[r * (long)C + c] : (_Float16)0.f;
}

__global__ void padvec_kernel(const float* __restrict__ v, float* __restrict__ o,
                              int n, int np) {
  int i = threadIdx.x;
  if (i < np) o[i] = (i < n) ? v[i] : 0.f;
}

// --- layernorm (one wave / row) with optional fused f16 output ------------
__global__ void ln_kernel(const float* __restrict__ x, const float* __restrict__ g,
                          const float* __restrict__ b, float* __restrict__ out,
                          _Float16* __restrict__ o16, int Dd) {
  long row = blockIdx.x;
  int lane = threadIdx.x;
  const float* xr = x + row * (long)Dd;
  const int per = Dd / 32;
  float vals[8], s = 0.f, s2 = 0.f;
  for (int t = 0; t < per; ++t) {
    float v = xr[lane + t * 32];
    vals[t] = v; s += v; s2 += v * v;
  }
  for (int o = 16; o; o >>= 1) { s += __shfl_xor(s, o); s2 += __shfl_xor(s2, o); }
  float mean = s / Dd;
  float inv  = rsqrtf(s2 / Dd - mean * mean + 1e-5f);
  for (int t = 0; t < per; ++t) {
    int c = lane + t * 32;
    float v = (vals[t] - mean) * inv * g[c] + b[c];
    out[row * (long)Dd + c] = v;
    if (o16) o16[row * (long)Dd + c] = (_Float16)v;
  }
}

// --- element-wise ---------------------------------------------------------
__global__ void add2_kernel(const float* a, const float* b, float* o, long n) {
  long i = (long)blockIdx.x * blockDim.x + threadIdx.x;
  if (i < n) o[i] = a[i] + b[i];
}
__global__ void addip_kernel(float* o, const float* a, _Float16* o16, long n) {
  long i = (long)blockIdx.x * blockDim.x + threadIdx.x;
  if (i < n) {
    float v = o[i] + a[i];
    o[i] = v;
    if (o16) o16[i] = (_Float16)v;
  }
}

// out[b,i,j,d] = act( (et?) + ri[b,i,d] + rj[b,j,d] + bias[d] ), f32/f16 outs optional
__global__ void edge_combine_kernel(const float* __restrict__ et,
                                    const float* __restrict__ ri,
                                    const float* __restrict__ rj,
                                    const float* __restrict__ bias,
                                    float* __restrict__ of32,
                                    _Float16* __restrict__ of16, int act) {
  long idx = (long)blockIdx.x * blockDim.x + threadIdx.x;
  if (idx >= kE * (long)kD) return;
  int d    = (int)(idx % kD);
  long row = idx / kD;
  int j    = (int)(row % kN);
  long bi  = row / kN;
  int b    = (int)(bi / kN);
  float v = ri[bi * kD + d] + rj[((long)b * kN + j) * kD + d];
  if (et)   v += et[idx];
  if (bias) v += bias[d];
  v = apply_act(v, act);
  if (of32) of32[idx] = v;
  if (of16) of16[idx] = (_Float16)v;
}

// --- spatial features: writes padded f16 rows of 32 directly --------------
__global__ void spatial_kernel(const float* __restrict__ boxes, _Float16* __restrict__ sf) {
  long idx = (long)blockIdx.x * blockDim.x + threadIdx.x;
  if (idx >= kE) return;
  int j = (int)(idx % kN);
  long t = idx / kN;
  int i = (int)(t % kN);
  int b = (int)(t / kN);
  const float* pi = boxes + ((long)b * kN + i) * 4;
  const float* pj = boxes + ((long)b * kN + j) * 4;
  float xi = pi[0], yi = pi[1], wi = pi[2], hi = pi[3];
  float xj = pj[0], yj = pj[1], wj = pj[2], hj = pj[3];
  float dx = (xj + wj * 0.5f) - (xi + wi * 0.5f);
  float dy = (yj + hj * 0.5f) - (yi + hi * 0.5f);
  float dist = sqrtf(dx * dx + dy * dy + EPSF);
  float ang = atan2f(dy, dx);
  float ai = wi * hi + EPSF, aj = wj * hj + EPSF;
  float aspi = fminf(fmaxf(wi / (hi + EPSF), 0.1f), 10.f);
  float aspj = fminf(fmaxf(wj / (hj + EPSF), 0.1f), 10.f);
  float iw = fmaxf(fminf(xi + wi, xj + wj) - fmaxf(xi, xj), 0.f);
  float ih = fmaxf(fminf(yi + hi, yj + hj) - fmaxf(yi, yj), 0.f);
  float inter = iw * ih;
  _Float16* o = sf + idx * 32;
  o[0] = (_Float16)dx;
  o[1] = (_Float16)dy;
  o[2] = (_Float16)dist;
  o[3] = (_Float16)sinf(ang);
  o[4] = (_Float16)cosf(ang);
  o[5] = (_Float16)logf(wj / (wi + EPSF) + EPSF);
  o[6] = (_Float16)logf(hj / (hi + EPSF) + EPSF);
  o[7] = (_Float16)logf(aj / (ai + EPSF) + EPSF);
  o[8] = (_Float16)logf(aspj / (aspi + EPSF) + EPSF);
  o[9] = (_Float16)(inter / (ai + aj - inter + EPSF));
  o[10] = (_Float16)(inter / (aj + EPSF));
  o[11] = (_Float16)(inter / (ai + EPSF));
  o[12] = (_Float16)logf(aspj + EPSF);
  o[13] = (_Float16)logf(aspi + EPSF);
  o[14] = (_Float16)1.f;
#pragma unroll
  for (int c = 15; c < 32; ++c) o[c] = (_Float16)0.f;
}

// --- attention softmax over j, then mean over heads -----------------------
__global__ void attn_stats_kernel(const float* __restrict__ a, float* __restrict__ st) {
  int id = blockIdx.x;              // bi*H + h
  int h  = id % kH;
  long bi = id / kH;
  int lane = threadIdx.x;
  const float* base = a + bi * kN * 16 + h;
  float v[4], mx = -1e30f;
#pragma unroll
  for (int t = 0; t < 4; ++t) { v[t] = base[(long)(lane + t * 32) * 16]; mx = fmaxf(mx, v[t]); }
  for (int o = 16; o; o >>= 1) mx = fmaxf(mx, __shfl_xor(mx, o));
  float s = 0.f;
#pragma unroll
  for (int t = 0; t < 4; ++t) s += expf(v[t] - mx);
  for (int o = 16; o; o >>= 1) s += __shfl_xor(s, o);
  if (lane == 0) { st[id * 2] = mx; st[id * 2 + 1] = s; }
}

__global__ void attn_mean_kernel(const float* __restrict__ a, const float* __restrict__ st,
                                 _Float16* __restrict__ at16) {
  long idx = (long)blockIdx.x * blockDim.x + threadIdx.x;
  if (idx >= kE) return;
  long bi = idx / kN;
  const float* ar = a + idx * 16;
  const float* s  = st + bi * kH * 2;
  float acc = 0.f;
#pragma unroll
  for (int h = 0; h < kH; ++h) acc += expf(ar[h] - s[2 * h]) / s[2 * h + 1];
  at16[idx] = (_Float16)(acc * (1.f / kH));
}

// --- edge += softmax(logits) @ type_emb, fused f16 out --------------------
__global__ void rel_emb_kernel(const float* __restrict__ logits,
                               const float* __restrict__ temb,
                               float* __restrict__ edge,
                               _Float16* __restrict__ e16) {
  long row = blockIdx.x;
  int d = threadIdx.x;
  const float* l = logits + row * kT;
  float mx = -1e30f;
#pragma unroll
  for (int t = 0; t < kT; ++t) mx = fmaxf(mx, l[t]);
  float e[kT], s = 0.f;
#pragma unroll
  for (int t = 0; t < kT; ++t) { e[t] = expf(l[t] - mx); s += e[t]; }
  float r = 0.f;
#pragma unroll
  for (int t = 0; t < kT; ++t) r += e[t] * temb[t * kD + d];
  float v = edge[row * (long)kD + d] + r / s;
  edge[row * (long)kD + d] = v;
  e16[row * (long)kD + d] = (_Float16)v;
}

// ===========================================================================
extern "C" void kernel_launch(void* const* d_in, const int* in_sizes, int n_in,
                              void* d_out, int out_size, void* d_ws, size_t ws_size,
                              hipStream_t stream) {
  (void)in_sizes; (void)n_in; (void)out_size; (void)ws_size;

  auto F = [&](int i) -> const float* { return (const float*)d_in[i]; };
  const float* obj   = F(0);
  const float* boxes = F(1);
  const int LBASE[3] = {4, 24, 44};
  enum { O_We1_W = 0, O_We1_b, O_We2_W, O_We2_b, O_Wm_W, O_Wm_b, O_Wn_W, O_Wn_b,
         O_Wo_W, O_Wo_b, O_Wu1_W, O_Wu1_b, O_Wu2_W, O_Wu2_b,
         O_en_b, O_en_g, O_ln_b, O_ln_g, O_nn_b, O_nn_g };
  const int I_eo_W = 2, I_eo_b = 3, I_no_W = 64, I_no_b = 65,
            I_rp1_W = 66, I_rp1_b = 67, I_rp2_W = 68, I_rp2_b = 69,
            I_rp3_W = 70, I_rp3_b = 71, I_sem1_W = 72, I_sem1_b = 73,
            I_sem2_W = 74, I_sem2_b = 75, I_sp1_W = 76, I_sp1_b = 77,
            I_sp2_W = 78, I_sp2_b = 79, I_sp_ln_b = 80, I_sp_ln_g = 81,
            I_type_emb = 82;

  char* wsp = (char*)d_ws;
  size_t off = 0;
  auto allocB = [&](size_t bytes) -> void* {
    void* p = wsp + off;
    off = (off + bytes + 255) & ~(size_t)255;
    return p;
  };
  float* t1    = (float*)allocB(kE * kD * 4);
  float* t2    = (float*)allocB(kE * kD * 4);
  float* edge  = (float*)allocB(kE * kD * 4);
  float* alog  = (float*)allocB(kE * 16 * 4);
  float* stats = (float*)allocB(kNB * kH * 2 * 4);
  float* node  = (float*)allocB(kNB * kD * 4);
  float* msgs  = (float*)allocB(kNB * kD * 4);
  float* nA    = (float*)allocB(kNB * kD * 4);
  float* nB    = (float*)allocB(kNB * kD * 4);
  float* nt    = (float*)allocB(kNB * kD * 4);
  float* nt2   = (float*)allocB(kNB * kD * 4);
  float* biasp = (float*)allocB(64);
  _Float16* e16  = (_Float16*)allocB(kE * kD * 2);
  _Float16* g16  = (_Float16*)allocB(kE * kD * 2);
  _Float16* sf16 = (_Float16*)allocB(kE * 32 * 2);
  _Float16* n16  = (_Float16*)allocB(kNB * kD * 2);
  _Float16* h16  = (_Float16*)allocB(kNB * kD * 2);
  _Float16* ag16 = (_Float16*)allocB(kNB * kD * 2);
  _Float16* at16 = (_Float16*)allocB(kE * 2);
  _Float16* m16s = (_Float16*)allocB(kBt * 32768 * 2);  // swizzled msgs, 4 batches
  _Float16* w16s = (_Float16*)allocB(768 * 256 * 2);
  _Float16* w16b = (_Float16*)allocB(256 * 16 * 2);

  float* out_nodes  = (float*)d_out;
  float* out_edge   = out_nodes + kNB * kD;
  float* out_logits = out_edge + kE * kD;

  auto GRID = [](long n, int b) { return dim3((unsigned)((n + b - 1) / b)); };
  auto CVT = [&](const float* s, _Float16* d, long rows, int C, int Cp) {
    cvtk_kernel<<<GRID(rows * (long)Cp, 256), 256, 0, stream>>>(s, d, rows, C, Cp);
  };
  auto CVTWS = [&](const float* s, _Float16* d, int R, int C, int Rp, int Cp,
                   int batch, long sSrc, long sDst) {
    long per = (long)(Rp >> 5) * (Cp >> 4) * 512;
    long total = per * batch;
    cvtws_kernel<<<GRID(total, 256), 256, 0, stream>>>(s, d, R, C, Rp, Cp, sSrc, sDst, total);
  };
  auto SLICE = [&](_Float16* w, int r0, int Cp) {   // row-slice of swizzled weight
    return w + (long)(r0 / 32) * (Cp / 16) * 512;
  };
  auto GEMMW = [&](const _Float16* A, const _Float16* Bs, const float* bias, float* C,
                   _Float16* C16, long M, int Nc, int K, int act,
                   int batch, long sA, long sB, long sC) {
    dim3 grid(Nc / 64, (unsigned)(M / 128), batch);
    gemm_wmma<2, 4><<<grid, 128, 0, stream>>>(A, Bs, bias, C, C16, Nc, K, sA, sB, sC, act);
  };
  auto GEMMN = [&](const _Float16* A, const _Float16* Bs, const float* bias, float* C,
                   _Float16* C16, long M, int K, int act) {   // Nc == 16
    dim3 grid(1, (unsigned)(M / 128), 1);
    gemm_wmma<2, 1><<<grid, 128, 0, stream>>>(A, Bs, bias, C, C16, 16, K, 0, 0, 0, act);
  };
  auto LN = [&](const float* x, const float* g, const float* b, float* o,
                _Float16* o16, long rows) {
    ln_kernel<<<dim3((unsigned)rows), 32, 0, stream>>>(x, g, b, o, o16, kD);
  };
  auto ADD2 = [&](const float* a, const float* b, float* o, long n) {
    add2_kernel<<<GRID(n, 256), 256, 0, stream>>>(a, b, o, n);
  };
  auto ADDIP = [&](float* o, const float* a, _Float16* o16, long n) {
    addip_kernel<<<GRID(n, 256), 256, 0, stream>>>(o, a, o16, n);
  };
  auto COMBINE = [&](const float* et, const float* ri, const float* rj,
                     const float* bias, float* of32, _Float16* of16, int act) {
    edge_combine_kernel<<<GRID(kE * (long)kD, 256), 256, 0, stream>>>(et, ri, rj, bias,
                                                                      of32, of16, act);
  };

  // ================= spatial edges =================
  spatial_kernel<<<GRID(kE, 256), 256, 0, stream>>>(boxes, sf16);
  CVTWS(F(I_sp1_W), w16s, 15, 128, 32, 128, 1, 0, 0);
  GEMMW(sf16, w16s, F(I_sp1_b), nullptr, e16, kE, 128, 32, ACT_GELU, 1, 0, 0, 0);
  CVTWS(F(I_sp2_W), w16s, 128, 256, 128, 256, 1, 0, 0);
  GEMMW(e16, w16s, F(I_sp2_b), t2, nullptr, kE, 256, 128, ACT_NONE, 1, 0, 0, 0);
  LN(t2, F(I_sp_ln_g), F(I_sp_ln_b), edge, nullptr, kE);

  // ================= semantic edges (concat split) =================
  CVT(obj, n16, kNB, 256, 256);
  CVTWS(F(I_sem1_W), w16s, 512, 256, 512, 256, 1, 0, 0);
  GEMMW(n16, w16s,                nullptr, nA, nullptr, kNB, 256, 256, ACT_NONE, 1, 0, 0, 0);
  GEMMW(n16, SLICE(w16s, 256, 256), nullptr, nB, nullptr, kNB, 256, 256, ACT_NONE, 1, 0, 0, 0);
  COMBINE(nullptr, nA, nB, F(I_sem1_b), nullptr, g16, ACT_GELU);
  CVTWS(F(I_sem2_W), w16s, 256, 256, 256, 256, 1, 0, 0);
  GEMMW(g16, w16s, F(I_sem2_b), t2, nullptr, kE, 256, 256, ACT_NONE, 1, 0, 0, 0);
  ADDIP(edge, t2, e16, kE * (long)kD);    // edge = spatial + sem (f32 + f16)

  (void)hipMemcpyAsync(node, obj, kNB * kD * sizeof(float),
                       hipMemcpyDeviceToDevice, stream);

  // ================= graph layers =================
  for (int l = 0; l < 3; ++l) {
    const int LB = LBASE[l];
    auto LP = [&](int o) -> const float* { return F(LB + o); };

    // h = node @ Wn + b  (f16 out only)
    CVTWS(LP(O_Wn_W), w16s, 256, 256, 256, 256, 1, 0, 0);
    GEMMW(n16, w16s, LP(O_Wn_b), nullptr, h16, kNB, 256, 256, ACT_NONE, 1, 0, 0, 0);

    // a = (leaky(ei@We1+b)) @ We2 + b, via concat split
    CVTWS(LP(O_We1_W), w16s, 768, 256, 768, 256, 1, 0, 0);
    GEMMW(h16, w16s,                nullptr, nA, nullptr, kNB, 256, 256, ACT_NONE, 1, 0, 0, 0);
    GEMMW(h16, SLICE(w16s, 256, 256), nullptr, nB, nullptr, kNB, 256, 256, ACT_NONE, 1, 0, 0, 0);
    GEMMW(e16, SLICE(w16s, 512, 256), nullptr, t1, nullptr, kE, 256, 256, ACT_NONE, 1, 0, 0, 0);
    COMBINE(t1, nA, nB, LP(O_We1_b), nullptr, g16, ACT_LEAKY);
    CVTWS(LP(O_We2_W), w16b, 256, 8, 256, 16, 1, 0, 0);
    padvec_kernel<<<1, 16, 0, stream>>>(LP(O_We2_b), biasp, 8, 16);
    GEMMN(g16, w16b, biasp, alog, nullptr, kE, 256, ACT_NONE);

    attn_stats_kernel<<<dim3((unsigned)(kNB * kH)), 32, 0, stream>>>(alog, stats);
    attn_mean_kernel<<<GRID(kE, 256), 256, 0, stream>>>(alog, stats, at16);

    // msgs = h @ Wm + b ; agg = attn @ msgs (batched, B swizzled)
    CVTWS(LP(O_Wm_W), w16s, 256, 256, 256, 256, 1, 0, 0);
    GEMMW(h16, w16s, LP(O_Wm_b), msgs, nullptr, kNB, 256, 256, ACT_NONE, 1, 0, 0, 0);
    CVTWS(msgs, m16s, 128, 256, 128, 256, kBt, 128L * 256, 32768);
    GEMMW(at16, m16s, nullptr, nullptr, ag16, 128, 256, 128, ACT_NONE,
          kBt, 128L * 128, 32768, 128L * 256);

    // node = ln(node + agg@Wo + b)
    CVTWS(LP(O_Wo_W), w16s, 256, 256, 256, 256, 1, 0, 0);
    GEMMW(ag16, w16s, LP(O_Wo_b), nt, nullptr, kNB, 256, 256, ACT_NONE, 1, 0, 0, 0);
    ADD2(node, nt, nt2, kNB * (long)kD);
    LN(nt2, LP(O_ln_g), LP(O_ln_b), node, n16, kNB);

    // u path (pre-update h16 / e16)
    CVTWS(LP(O_Wu1_W), w16s, 768, 256, 768, 256, 1, 0, 0);
    GEMMW(h16, w16s,                nullptr, nA, nullptr, kNB, 256, 256, ACT_NONE, 1, 0, 0, 0);
    GEMMW(h16, SLICE(w16s, 256, 256), nullptr, nB, nullptr, kNB, 256, 256, ACT_NONE, 1, 0, 0, 0);
    GEMMW(e16, SLICE(w16s, 512, 256), nullptr, t1, nullptr, kE, 256, 256, ACT_NONE, 1, 0, 0, 0);
    COMBINE(t1, nA, nB, LP(O_Wu1_b), nullptr, g16, ACT_GELU);
    CVTWS(LP(O_Wu2_W), w16s, 256, 256, 256, 256, 1, 0, 0);
    GEMMW(g16, w16s, LP(O_Wu2_b), t1, nullptr, kE, 256, 256, ACT_NONE, 1, 0, 0, 0);
    ADDIP(edge, t1, nullptr, kE * (long)kD);

    // post-layer norms (fused f16 outs feed next layer / head)
    LN(node, LP(O_nn_g), LP(O_nn_b), node, n16, kNB);
    LN(edge, LP(O_en_g), LP(O_en_b), edge, e16, kE);
  }

  // ================= relation head =================
  CVTWS(F(I_rp1_W), w16s, 256, 256, 256, 256, 1, 0, 0);
  GEMMW(e16, w16s, F(I_rp1_b), nullptr, g16, kE, 256, 256, ACT_GELU, 1, 0, 0, 0);
  CVTWS(F(I_rp2_W), w16s, 256, 128, 256, 128, 1, 0, 0);
  GEMMW(g16, w16s, F(I_rp2_b), nullptr, e16, kE, 128, 256, ACT_GELU, 1, 0, 0, 0);
  CVTWS(F(I_rp3_W), w16b, 128, 16, 128, 16, 1, 0, 0);
  GEMMN(e16, w16b, F(I_rp3_b), out_logits, nullptr, kE, 128, ACT_NONE);

  // edge += softmax(logits) @ type_emb (rewrites e16 fully)
  rel_emb_kernel<<<dim3((unsigned)kE), 256, 0, stream>>>(out_logits, F(I_type_emb), edge, e16);

  // ================= output projections =================
  CVTWS(F(I_no_W), w16s, 256, 256, 256, 256, 1, 0, 0);
  GEMMW(n16, w16s, F(I_no_b), out_nodes, nullptr, kNB, 256, 256, ACT_NONE, 1, 0, 0, 0);
  CVTWS(F(I_eo_W), w16s, 256, 256, 256, 256, 1, 0, 0);
  GEMMW(e16, w16s, F(I_eo_b), out_edge, nullptr, kE, 256, 256, ACT_NONE, 1, 0, 0, 0);
}
